// OldEncoderLayer_75350906241671
// MI455X (gfx1250) — compile-verified, matrix operand order checked by reading, not run on hardware
//
#include <hip/hip_runtime.h>

typedef __bf16 bf16_t;
typedef __attribute__((ext_vector_type(16))) __bf16 v16bf;
typedef __attribute__((ext_vector_type(8)))  __bf16 v8bf;
typedef __attribute__((ext_vector_type(8)))  float  v8f;

#define N_NODES 16384
#define KNB 48
#define CV 128
#define CE 128
#define HID 128
#define INW 384      // CV + CE + CV
#define DWID 512     // dense hidden
#define LN_EPS 1e-5f

// ---------------- device helpers ----------------

__device__ __forceinline__ float gelu_exact(float x) {
    return 0.5f * x * (1.0f + erff(x * 0.7071067811865475f));
}

__device__ __forceinline__ float warp_sum(float x) {
#pragma unroll
    for (int o = 16; o > 0; o >>= 1) x += __shfl_xor(x, o, 32);
    return x;
}

// CDNA5 async global->LDS copy (16B per lane), tracked on ASYNCcnt.
// lds_off = LDS byte address (low 32 bits of generic shared pointer,
// per ISA flat-aperture rule).
__device__ __forceinline__ void async_copy_b128(uint32_t lds_off,
                                                const void* gptr) {
    asm volatile("global_load_async_to_lds_b128 %0, %1, off"
                 :: "v"(lds_off), "v"((unsigned long long)(uintptr_t)gptr)
                 : "memory");
}

__device__ __forceinline__ void wait_async0() {
    asm volatile("s_wait_asynccnt 0x0" ::: "memory");
}

__device__ __forceinline__ uint32_t lds_addr32(const void* p) {
    return (uint32_t)(uintptr_t)p;
}

// Build a 16-element bf16 fragment from two contiguous 16B segments.
// Per CDNA5 16-bit A/B layout: lane half=lane>>4 selects K in [0..7]+[16..23]
// (half 0) or [8..15]+[24..31] (half 1) within a K=32 chunk.
__device__ __forceinline__ v16bf make_frag(const bf16_t* p) {
    v8bf lo = *(const v8bf*)p;        // K = kc*32 + half*8 .. +7
    v8bf hi = *(const v8bf*)(p + 16); // K = kc*32 + 16 + half*8 .. +7
    v16bf r;
#pragma unroll
    for (int i = 0; i < 8; i++) { r[i] = lo[i]; r[i + 8] = hi[i]; }
    return r;
}

// A fragment: 16x32 tile of row-major bf16 activations (ldw elems/row).
__device__ __forceinline__ v16bf load_a(const bf16_t* act, int ldw, int mtile,
                                        int kc, int lane) {
    int row  = mtile * 16 + (lane & 15);
    int half = lane >> 4;
    return make_frag(act + row * ldw + kc * 32 + half * 8);
}

// B fragment: 32x16 tile; W row-major (out,in); tile column = out index.
__device__ __forceinline__ v16bf load_b(const bf16_t* W, int in_w, int ntile,
                                        int kc, int lane) {
    int col  = ntile * 16 + (lane & 15);
    int half = lane >> 4;
    return make_frag(W + (size_t)col * in_w + kc * 32 + half * 8);
}

__device__ __forceinline__ v8f wmma_bf16(v16bf a, v16bf b, v8f c) {
    return __builtin_amdgcn_wmma_f32_16x16x32_bf16(false, a, false, b,
                                                   (short)0, c, false, false);
}

// One MLP layer (bias + exact GELU), writing bf16 to LDS.
// C/D layout: VGPR v, lane l -> row v + 8*(l>=16), col l%16.
template <int MTI>
__device__ __forceinline__ void mlp_layer_gelu(const bf16_t* act_in, int ldw_in,
                                               int kchunks, const bf16_t* W,
                                               int in_w, const float* bias,
                                               bf16_t* act_out, int ldw_out,
                                               int ntile, int lane) {
    const int half = lane >> 4, colL = lane & 15;
    v8f acc[MTI];
#pragma unroll
    for (int m = 0; m < MTI; m++) acc[m] = {};
    for (int kc = 0; kc < kchunks; kc++) {
        v16bf B = load_b(W, in_w, ntile, kc, lane);
#pragma unroll
        for (int m = 0; m < MTI; m++) {
            v16bf A = load_a(act_in, ldw_in, m, kc, lane);
            acc[m]  = wmma_bf16(A, B, acc[m]);
        }
    }
    float bs = bias[ntile * 16 + colL];
#pragma unroll
    for (int m = 0; m < MTI; m++) {
#pragma unroll
        for (int v = 0; v < 8; v++) {
            int row = m * 16 + v + half * 8;
            act_out[row * ldw_out + ntile * 16 + colL] =
                (bf16_t)gelu_exact(acc[m][v] + bs);
        }
    }
}

// ---------------- kernels ----------------

__global__ void f32_to_bf16_kernel(const float* __restrict__ src,
                                   bf16_t* __restrict__ dst, int n) {
    int i = blockIdx.x * blockDim.x + threadIdx.x;
    if (i < n) dst[i] = (bf16_t)src[i];
}

// ---- edge MLP: per block NG nodes (NG*48 edge rows), 8 waves, each wave
// owns a 16-col strip of each 128-wide layer. mode 0: masked sum over K /30.
// mode 1: per-edge residual + LayerNorm, write h_E output.
#define NG 2
#define EROWS (NG * KNB)       // 96
#define EMT (EROWS / 16)       // 6
#define LDW_IN (INW + 8)       // 392 (pad to break LDS bank stride)
#define LDW_H (HID + 8)        // 136

__global__ void __launch_bounds__(256) edge_mlp_kernel(
    const bf16_t* __restrict__ hv_bf, const float* __restrict__ h_E,
    const int* __restrict__ nbr,
    const bf16_t* __restrict__ w1, const float* __restrict__ b1,
    const bf16_t* __restrict__ w2, const float* __restrict__ b2,
    const bf16_t* __restrict__ w3, const float* __restrict__ b3,
    const float* __restrict__ mask_att, float* __restrict__ dh_out,
    const float* __restrict__ ln_w, const float* __restrict__ ln_b,
    float* __restrict__ hE_out, int mode) {
    __shared__ __align__(16) char smem[EROWS * LDW_IN * 2 + 2 * EROWS * LDW_H * 2];
    bf16_t* in_act = (bf16_t*)smem;
    bf16_t* act1   = (bf16_t*)(smem + EROWS * LDW_IN * 2);
    bf16_t* act2   = act1 + EROWS * LDW_H;
    float*  fbuf   = (float*)smem;  // aliases in_act (dead after layer 1)

    const int tid  = threadIdx.x;
    const int lane = tid & 31;
    const int wv   = tid >> 5;
    const int half = lane >> 4;
    const int colL = lane & 15;
    const int n0   = blockIdx.x * NG;

    // ---- stage h_EV = [h_self | h_E | h_nb] as bf16 ----
    // self + gathered-neighbor columns are already bf16 in global: use the
    // CDNA5 async global->LDS path (no VGPR round trip, tracked on ASYNCcnt).
    const uint32_t lds_in = lds_addr32(in_act);
    for (int idx = tid; idx < EROWS * 32; idx += 256) {
        int row = idx >> 5, ch = idx & 31;
        int n = n0 + row / KNB, k = row % KNB;
        int src_n = (ch < 16) ? n : nbr[n * KNB + k];
        int csrc  = (ch & 15) * 8;
        int cdst  = (ch < 16) ? csrc : (CV + CE + csrc);
        async_copy_b128(lds_in + (uint32_t)(row * LDW_IN + cdst) * 2,
                        hv_bf + (size_t)src_n * CV + csrc);
    }
    // middle 128 cols: f32 edge features -> bf16 via VALU
    for (int idx = tid; idx < EROWS * 16; idx += 256) {
        int row = idx >> 4;
        int c   = CV + (idx & 15) * 8;
        int n = n0 + row / KNB, k = row % KNB;
        const float* src = h_E + ((size_t)n * KNB + k) * CE + (c - CV);
        float4 f0 = *(const float4*)src;
        float4 f1 = *(const float4*)(src + 4);
        v8bf v;
        v[0] = (bf16_t)f0.x; v[1] = (bf16_t)f0.y;
        v[2] = (bf16_t)f0.z; v[3] = (bf16_t)f0.w;
        v[4] = (bf16_t)f1.x; v[5] = (bf16_t)f1.y;
        v[6] = (bf16_t)f1.z; v[7] = (bf16_t)f1.w;
        *(v8bf*)(in_act + row * LDW_IN + c) = v;
    }
    wait_async0();
    __syncthreads();

    mlp_layer_gelu<EMT>(in_act, LDW_IN, INW / 32, w1, INW, b1, act1, LDW_H, wv, lane);
    __syncthreads();
    mlp_layer_gelu<EMT>(act1, LDW_H, HID / 32, w2, HID, b2, act2, LDW_H, wv, lane);
    __syncthreads();

    // layer 3 (no activation)
    v8f acc3[EMT];
#pragma unroll
    for (int m = 0; m < EMT; m++) acc3[m] = {};
    for (int kc = 0; kc < HID / 32; kc++) {
        v16bf B = load_b(w3, HID, wv, kc, lane);
#pragma unroll
        for (int m = 0; m < EMT; m++) {
            v16bf A = load_a(act2, LDW_H, m, kc, lane);
            acc3[m] = wmma_bf16(A, B, acc3[m]);
        }
    }
    float b3c = b3[wv * 16 + colL];

    if (mode == 0) {
#pragma unroll
        for (int m = 0; m < EMT; m++) {
#pragma unroll
            for (int v = 0; v < 8; v++) {
                int row = m * 16 + v + half * 8;
                int n = n0 + row / KNB, k = row % KNB;
                float ma = mask_att[n * KNB + k];
                fbuf[row * CE + wv * 16 + colL] = (acc3[m][v] + b3c) * ma;
            }
        }
        __syncthreads();
        for (int idx = tid; idx < NG * CE; idx += 256) {
            int g = idx >> 7, c = idx & 127;
            float s = 0.f;
            for (int k = 0; k < KNB; k++) s += fbuf[(g * KNB + k) * CE + c];
            dh_out[(size_t)(n0 + g) * CE + c] = s * (1.0f / 30.0f);
        }
    } else {
#pragma unroll
        for (int m = 0; m < EMT; m++) {
#pragma unroll
            for (int v = 0; v < 8; v++) {
                int row = m * 16 + v + half * 8;
                fbuf[row * CE + wv * 16 + colL] = acc3[m][v] + b3c;
            }
        }
        __syncthreads();
        // LayerNorm per edge row; EROWS/8 rows per wave
        for (int rl = 0; rl < EROWS / 8; rl++) {
            int row = wv * (EROWS / 8) + rl;
            int n = n0 + row / KNB, k = row % KNB;
            size_t gbase = ((size_t)n * KNB + k) * CE;
            float x[4], s = 0.f, sq = 0.f;
#pragma unroll
            for (int j = 0; j < 4; j++) {
                int c = lane * 4 + j;
                x[j] = fbuf[row * CE + c] + h_E[gbase + c];
                s += x[j]; sq += x[j] * x[j];
            }
            s = warp_sum(s); sq = warp_sum(sq);
            float mean = s * (1.0f / CE);
            float var  = sq * (1.0f / CE) - mean * mean;
            float rstd = rsqrtf(var + LN_EPS);
#pragma unroll
            for (int j = 0; j < 4; j++) {
                int c = lane * 4 + j;
                hE_out[gbase + c] = (x[j] - mean) * rstd * ln_w[c] + ln_b[c];
            }
        }
    }
}

// ---- h_V1 = LN(h_V + dh) ; one row per wave
__global__ void __launch_bounds__(256) ln_update_kernel(
    const float* __restrict__ hv, const float* __restrict__ dh,
    const float* __restrict__ ln_w, const float* __restrict__ ln_b,
    float* __restrict__ out_f32, bf16_t* __restrict__ out_bf) {
    int wv = threadIdx.x >> 5, lane = threadIdx.x & 31;
    size_t n = (size_t)blockIdx.x * 8 + wv;
    float x[4], s = 0.f, sq = 0.f;
#pragma unroll
    for (int j = 0; j < 4; j++) {
        int c = lane * 4 + j;
        x[j] = hv[n * CV + c] + dh[n * CV + c];
        s += x[j]; sq += x[j] * x[j];
    }
    s = warp_sum(s); sq = warp_sum(sq);
    float mean = s * (1.0f / CV);
    float var  = sq * (1.0f / CV) - mean * mean;
    float rstd = rsqrtf(var + LN_EPS);
#pragma unroll
    for (int j = 0; j < 4; j++) {
        int c = lane * 4 + j;
        float y = (x[j] - mean) * rstd * ln_w[c] + ln_b[c];
        out_f32[n * CV + c] = y;
        out_bf[n * CV + c]  = (bf16_t)y;
    }
}

// ---- position-wise dense: 128 -> 512 (GELU) -> 128, residual + LN2 + mask
#define DM 64               // rows per block
#define DMT (DM / 16)       // 4
#define LDW1 (DWID + 8)     // 520

__global__ void __launch_bounds__(256) dense_kernel(
    const bf16_t* __restrict__ hv1_bf, const float* __restrict__ hv1_f32,
    const bf16_t* __restrict__ w1, const float* __restrict__ b1,
    const bf16_t* __restrict__ w2, const float* __restrict__ b2,
    const float* __restrict__ ln_w, const float* __restrict__ ln_b,
    const float* __restrict__ mask, float* __restrict__ hv2_out,
    bf16_t* __restrict__ hv2_bf) {
    __shared__ __align__(16) char smem[DM * LDW_H * 2 + DM * LDW1 * 2 + DM * CV * 4];
    bf16_t* inA  = (bf16_t*)smem;
    bf16_t* act1 = (bf16_t*)(smem + DM * LDW_H * 2);
    float*  fbuf = (float*)(smem + DM * LDW_H * 2 + DM * LDW1 * 2);

    const int tid  = threadIdx.x;
    const int lane = tid & 31;
    const int wv   = tid >> 5;
    const int colL = lane & 15;
    const int n0   = blockIdx.x * DM;

    // async-stage the bf16 input rows
    const uint32_t lds_in = lds_addr32(inA);
    for (int idx = tid; idx < DM * (CV / 8); idx += 256) {
        int row = idx >> 4, cb = idx & 15;
        async_copy_b128(lds_in + (uint32_t)(row * LDW_H + cb * 8) * 2,
                        hv1_bf + (size_t)(n0 + row) * CV + cb * 8);
    }
    wait_async0();
    __syncthreads();

    // layer 1: 512 cols = 32 ntiles; wave handles nt = wv + 8*j
    for (int j = 0; j < 4; j++) {
        mlp_layer_gelu<DMT>(inA, LDW_H, CV / 32, w1, CV, b1, act1, LDW1,
                            wv + 8 * j, lane);
    }
    __syncthreads();

    // layer 2: K = 512 (16 chunks), wave owns ntile wv
    {
        const int half = lane >> 4;
        v8f acc[DMT];
#pragma unroll
        for (int m = 0; m < DMT; m++) acc[m] = {};
        for (int kc = 0; kc < DWID / 32; kc++) {
            v16bf B = load_b(w2, DWID, wv, kc, lane);
#pragma unroll
            for (int m = 0; m < DMT; m++) {
                v16bf A = load_a(act1, LDW1, m, kc, lane);
                acc[m]  = wmma_bf16(A, B, acc[m]);
            }
        }
        float bs = b2[wv * 16 + colL];
#pragma unroll
        for (int m = 0; m < DMT; m++) {
#pragma unroll
            for (int v = 0; v < 8; v++) {
                int row = m * 16 + v + half * 8;
                fbuf[row * CV + wv * 16 + colL] = acc[m][v] + bs;
            }
        }
    }
    __syncthreads();

    // residual + LN2 + mask; 8 rows per wave
    for (int rl = 0; rl < DM / 8; rl++) {
        int r = wv * (DM / 8) + rl;
        size_t n = (size_t)n0 + r;
        float x[4], s = 0.f, sq = 0.f;
#pragma unroll
        for (int j = 0; j < 4; j++) {
            int c = lane * 4 + j;
            x[j] = hv1_f32[n * CV + c] + fbuf[r * CV + c];
            s += x[j]; sq += x[j] * x[j];
        }
        s = warp_sum(s); sq = warp_sum(sq);
        float mean = s * (1.0f / CV);
        float var  = sq * (1.0f / CV) - mean * mean;
        float rstd = rsqrtf(var + LN_EPS);
        float mk = mask[n];
#pragma unroll
        for (int j = 0; j < 4; j++) {
            int c = lane * 4 + j;
            float y = ((x[j] - mean) * rstd * ln_w[c] + ln_b[c]) * mk;
            hv2_out[n * CV + c] = y;
            hv2_bf[n * CV + c]  = (bf16_t)y;
        }
    }
}

// ---------------- host launch ----------------

extern "C" void kernel_launch(void* const* d_in, const int* in_sizes, int n_in,
                              void* d_out, int out_size, void* d_ws,
                              size_t ws_size, hipStream_t stream) {
    (void)in_sizes; (void)n_in; (void)out_size; (void)ws_size;

    const float* node_f = (const float*)d_in[0];
    const float* edge_f = (const float*)d_in[1];
    const int*   nbr    = (const int*)d_in[2];
    const float* mask   = (const float*)d_in[3];
    const float* mask_a = (const float*)d_in[4];
    const float* msg_w1 = (const float*)d_in[5];
    const float* msg_b1 = (const float*)d_in[6];
    const float* msg_w2 = (const float*)d_in[7];
    const float* msg_b2 = (const float*)d_in[8];
    const float* msg_w3 = (const float*)d_in[9];
    const float* msg_b3 = (const float*)d_in[10];
    const float* ln1_w  = (const float*)d_in[11];
    const float* ln1_b  = (const float*)d_in[12];
    const float* d_w1   = (const float*)d_in[13];
    const float* d_b1   = (const float*)d_in[14];
    const float* d_w2   = (const float*)d_in[15];
    const float* d_b2   = (const float*)d_in[16];
    const float* ln2_w  = (const float*)d_in[17];
    const float* ln2_b  = (const float*)d_in[18];
    const float* eu_w1  = (const float*)d_in[19];
    const float* eu_b1  = (const float*)d_in[20];
    const float* eu_w2  = (const float*)d_in[21];
    const float* eu_b2  = (const float*)d_in[22];
    const float* eu_w3  = (const float*)d_in[23];
    const float* eu_b3  = (const float*)d_in[24];
    const float* ln3_w  = (const float*)d_in[25];
    const float* ln3_b  = (const float*)d_in[26];

    char* ws = (char*)d_ws;
    const size_t SZ_WIN = 128 * 384, SZ_WH = 128 * 128, SZ_WD = 512 * 128;
    const size_t SZ_HV = (size_t)N_NODES * CV;
    size_t off = 0;
    auto take = [&](size_t bytes) { size_t o = off; off += (bytes + 255) & ~(size_t)255; return o; };
    bf16_t* wb_m1 = (bf16_t*)(ws + take(SZ_WIN * 2));
    bf16_t* wb_m2 = (bf16_t*)(ws + take(SZ_WH * 2));
    bf16_t* wb_m3 = (bf16_t*)(ws + take(SZ_WH * 2));
    bf16_t* wb_e1 = (bf16_t*)(ws + take(SZ_WIN * 2));
    bf16_t* wb_e2 = (bf16_t*)(ws + take(SZ_WH * 2));
    bf16_t* wb_e3 = (bf16_t*)(ws + take(SZ_WH * 2));
    bf16_t* wb_d1 = (bf16_t*)(ws + take(SZ_WD * 2));
    bf16_t* wb_d2 = (bf16_t*)(ws + take(SZ_WD * 2));
    bf16_t* hv0_bf  = (bf16_t*)(ws + take(SZ_HV * 2));
    bf16_t* hv1_bf  = (bf16_t*)(ws + take(SZ_HV * 2));
    bf16_t* hv2_bf  = (bf16_t*)(ws + take(SZ_HV * 2));
    float*  hv1_f32 = (float*)(ws + take(SZ_HV * 4));
    float*  dh_msg  = (float*)(ws + take(SZ_HV * 4));

    auto cvt = [&](const float* src, bf16_t* dst, size_t n) {
        f32_to_bf16_kernel<<<(unsigned)((n + 255) / 256), 256, 0, stream>>>(src, dst, (int)n);
    };
    cvt(msg_w1, wb_m1, SZ_WIN);
    cvt(msg_w2, wb_m2, SZ_WH);
    cvt(msg_w3, wb_m3, SZ_WH);
    cvt(eu_w1, wb_e1, SZ_WIN);
    cvt(eu_w2, wb_e2, SZ_WH);
    cvt(eu_w3, wb_e3, SZ_WH);
    cvt(d_w1, wb_d1, SZ_WD);
    cvt(d_w2, wb_d2, SZ_WD);
    cvt(node_f, hv0_bf, SZ_HV);

    float* out_hv = (float*)d_out;
    float* out_he = (float*)d_out + (size_t)N_NODES * CV;

    // message pass -> dh_msg
    edge_mlp_kernel<<<N_NODES / NG, 256, 0, stream>>>(
        hv0_bf, edge_f, nbr, wb_m1, msg_b1, wb_m2, msg_b2, wb_m3, msg_b3,
        mask_a, dh_msg, nullptr, nullptr, nullptr, 0);

    // h_V1 = LN1(h_V + dh)
    ln_update_kernel<<<N_NODES / 8, 256, 0, stream>>>(node_f, dh_msg, ln1_w,
                                                      ln1_b, hv1_f32, hv1_bf);

    // dense + LN2 + mask -> h_V output (and bf16 copy)
    dense_kernel<<<N_NODES / DM, 256, 0, stream>>>(
        hv1_bf, hv1_f32, wb_d1, d_b1, wb_d2, d_b2, ln2_w, ln2_b, mask, out_hv,
        hv2_bf);

    // edge update + LN3 -> h_E output
    edge_mlp_kernel<<<N_NODES / NG, 256, 0, stream>>>(
        hv2_bf, edge_f, nbr, wb_e1, eu_b1, wb_e2, eu_b2, wb_e3, eu_b3, nullptr,
        nullptr, ln3_w, ln3_b, out_he, 1);
}
